// Block_91061896609903
// MI455X (gfx1250) — matrix operand order repaired and checked by reading, not compiled, not run on hardware
//
#include <hip/hip_runtime.h>

// ---------------------------------------------------------------------------
// Transformer block (B=4, T=2048, C=1024, H=4096) on gfx1250 via WMMA bf16.
// GEMMs: v_wmma_f32_16x16x32_bf16, LDS-staged tiles via async global->LDS
// copies (ASYNCcnt) with double buffering.
// ---------------------------------------------------------------------------

#define CDIM 1024
#define TDIM 2048
#define BDIM 4
#define HDIM 4096
#define BT   (BDIM * TDIM)   // 8192

typedef __attribute__((ext_vector_type(16))) __bf16 v16bf;
typedef __attribute__((ext_vector_type(8)))  __bf16 v8bf;
typedef __attribute__((ext_vector_type(8)))  float  v8f;
typedef __attribute__((ext_vector_type(4)))  int    v4i;

union Frag16 { v16bf v; v8bf h[2]; };

__device__ __forceinline__ float gelu_f(float x) {
    const float c = 0.7978845608028654f; // sqrt(2/pi)
    float x3 = x * x * x;
    return 0.5f * x * (1.0f + tanhf(c * (x + 0.044715f * x3)));
}

// 16-byte global -> LDS copy. Prefer CDNA5 async-to-LDS (ASYNCcnt) when the
// toolchain exposes the builtin; otherwise plain load+store staging.
// Builtin signature (from probe diagnostics): (int4 AS1*, int4 AS3*, imm, imm).
__device__ __forceinline__ void cp_async16(const __bf16* g, __bf16* l) {
#if defined(__gfx1250__) && __has_builtin(__builtin_amdgcn_global_load_async_to_lds_b128)
    __builtin_amdgcn_global_load_async_to_lds_b128(
        (__attribute__((address_space(1))) v4i*)(const void*)g,
        (__attribute__((address_space(3))) v4i*)(void*)l, 0, 0);
#else
    *(v8bf*)l = *(const v8bf*)g;
#endif
}

__device__ __forceinline__ void wait_async() {
#if defined(__gfx1250__) && __has_builtin(__builtin_amdgcn_s_wait_asynccnt)
    __builtin_amdgcn_s_wait_asynccnt(0);
#endif
}

// ---------------------------------------------------------------------------
// Weight convert + transpose: wT[n*K + k] = (bf16) w[k*N + n]
// ---------------------------------------------------------------------------
__global__ void wt_kernel(const float* __restrict__ w, __bf16* __restrict__ wT,
                          int K, int N) {
    size_t idx = (size_t)blockIdx.x * 256 + threadIdx.x;
    if (idx >= (size_t)K * N) return;
    int k = (int)(idx / N);
    int n = (int)(idx % N);
    wT[(size_t)n * K + k] = (__bf16)w[idx];
}

// ---------------------------------------------------------------------------
// LayerNorm over last dim (cols), one block per row, bf16 output.
// ---------------------------------------------------------------------------
__global__ void ln_kernel(const float* __restrict__ x, const float* __restrict__ g,
                          const float* __restrict__ b, __bf16* __restrict__ out,
                          int cols) {
    int row = blockIdx.x;
    const float* xr = x + (size_t)row * cols;
    __shared__ float sred[256];
    int tid = threadIdx.x;

    float lsum = 0.0f;
    for (int c = tid; c < cols; c += 256) lsum += xr[c];
    sred[tid] = lsum; __syncthreads();
    for (int s = 128; s > 0; s >>= 1) {
        if (tid < s) sred[tid] += sred[tid + s];
        __syncthreads();
    }
    float mu = sred[0] / cols;
    __syncthreads();

    float lvar = 0.0f;
    for (int c = tid; c < cols; c += 256) { float d = xr[c] - mu; lvar += d * d; }
    sred[tid] = lvar; __syncthreads();
    for (int s = 128; s > 0; s >>= 1) {
        if (tid < s) sred[tid] += sred[tid + s];
        __syncthreads();
    }
    float rstd = rsqrtf(sred[0] / cols + 1e-5f);
    __syncthreads();

    __bf16* orow = out + (size_t)row * cols;
    for (int c = tid; c < cols; c += 256)
        orow[c] = (__bf16)((xr[c] - mu) * rstd * g[c] + b[c]);
}

// ---------------------------------------------------------------------------
// Split qkv f32 [BT, 3C] -> q bf16 [BT,C] (scaled 1/sqrt(C)), k bf16 [BT,C],
// vT bf16 per-batch [C,T].
// ---------------------------------------------------------------------------
__global__ void qkv_split_kernel(const float* __restrict__ qkv,
                                 __bf16* __restrict__ q, __bf16* __restrict__ k,
                                 __bf16* __restrict__ vT) {
    size_t idx = (size_t)blockIdx.x * 256 + threadIdx.x;
    if (idx >= (size_t)BT * 3 * CDIM) return;
    int m = (int)(idx / (3 * CDIM));
    int n = (int)(idx % (3 * CDIM));
    float v = qkv[idx];
    if (n < CDIM) {
        q[(size_t)m * CDIM + n] = (__bf16)(v * 0.03125f); // 1/sqrt(1024)
    } else if (n < 2 * CDIM) {
        k[(size_t)m * CDIM + (n - CDIM)] = (__bf16)v;
    } else {
        int bb = m >> 11;          // m / T
        int t  = m & (TDIM - 1);   // m % T
        int c  = n - 2 * CDIM;
        vT[((size_t)bb * CDIM + c) * TDIM + t] = (__bf16)v;
    }
}

// ---------------------------------------------------------------------------
// Row softmax over length `cols` of f32, writes bf16. One block per row.
// ---------------------------------------------------------------------------
__global__ void softmax_kernel(const float* __restrict__ att,
                               __bf16* __restrict__ out, int cols) {
    int row = blockIdx.x;
    const float* ar = att + (size_t)row * cols;
    __shared__ float sred[256];
    int tid = threadIdx.x;

    float lmax = -3.4e38f;
    for (int c = tid; c < cols; c += 256) lmax = fmaxf(lmax, ar[c]);
    sred[tid] = lmax; __syncthreads();
    for (int s = 128; s > 0; s >>= 1) {
        if (tid < s) sred[tid] = fmaxf(sred[tid], sred[tid + s]);
        __syncthreads();
    }
    float mx = sred[0];
    __syncthreads();

    float lsum = 0.0f;
    for (int c = tid; c < cols; c += 256) lsum += __expf(ar[c] - mx);
    sred[tid] = lsum; __syncthreads();
    for (int s = 128; s > 0; s >>= 1) {
        if (tid < s) sred[tid] += sred[tid + s];
        __syncthreads();
    }
    float inv = 1.0f / sred[0];
    __syncthreads();

    __bf16* orow = out + (size_t)row * cols;
    for (int c = tid; c < cols; c += 256)
        orow[c] = (__bf16)(__expf(ar[c] - mx) * inv);
}

// ---------------------------------------------------------------------------
// NT GEMM via WMMA bf16: D[M,N] = A[M,K] @ Bt[N,K]^T  (+ epilogue)
//   MODE 0: Df = acc                       (attention scores)
//   MODE 1: Db = bf16(acc)                 (att @ V)
//   MODE 2: Df = acc + bias + resid        (proj / final, f32 out)
//   MODE 3: Db = bf16(gelu(acc + bias))    (fc)
//   MODE 4: Df = acc + bias                (qkv)
// 256 threads = 8 wave32; block tile 128(M) x 128(N); wave tile 32x64.
// K-step 32, double-buffered LDS tiles filled with async global->LDS copies.
// ---------------------------------------------------------------------------
template <int MODE>
__global__ __launch_bounds__(256) void gemm_nt(
    const __bf16* __restrict__ A, const __bf16* __restrict__ Bt,
    float* __restrict__ Df, __bf16* __restrict__ Db,
    const float* __restrict__ bias, const float* __restrict__ resid,
    int M, int N, int K,
    long long sA, long long sB, long long sD)
{
    __shared__ __align__(16) __bf16 sAt[2][128 * 32];
    __shared__ __align__(16) __bf16 sBt[2][128 * 32];

    const int z = blockIdx.z;
    A  += (size_t)z * sA;
    Bt += (size_t)z * sB;
    const size_t dOff = (size_t)z * sD;

    const int tid  = threadIdx.x;
    const int lane = tid & 31;
    const int wave = tid >> 5;
    const int wmBase = (wave >> 1) * 32;   // 0,32,64,96
    const int wnBase = (wave & 1) * 64;    // 0,64
    const int m0 = blockIdx.y * 128;
    const int n0 = blockIdx.x * 128;

    const int lid = lane & 15;   // row (A) / col (B) within a 16-tile
    const int kg  = lane >> 4;   // K-group select

    // staging coordinates: 128 rows x 32 bf16 = 512 x 16B chunks, 2 per thread
    const int c0r = tid >> 2,          c0k = (tid & 3) * 8;
    const int c1r = (tid + 256) >> 2,  c1k = ((tid + 256) & 3) * 8;

    v8f acc[2][4];
    #pragma unroll
    for (int ti = 0; ti < 2; ++ti)
        #pragma unroll
        for (int tj = 0; tj < 4; ++tj) acc[ti][tj] = {};

    const __bf16* gA = A + (size_t)m0 * K;
    const __bf16* gB = Bt + (size_t)n0 * K;

    // prologue: stage K-step 0 into buffer 0
    cp_async16(gA + (size_t)c0r * K + c0k, &sAt[0][c0r * 32 + c0k]);
    cp_async16(gA + (size_t)c1r * K + c1k, &sAt[0][c1r * 32 + c1k]);
    cp_async16(gB + (size_t)c0r * K + c0k, &sBt[0][c0r * 32 + c0k]);
    cp_async16(gB + (size_t)c1r * K + c1k, &sBt[0][c1r * 32 + c1k]);
    wait_async();
    __syncthreads();

    int buf = 0;
    for (int k0 = 0; k0 < K; k0 += 32) {
        // stage next K-step into the other buffer
        if (k0 + 32 < K) {
            const __bf16* gAn = gA + k0 + 32;
            const __bf16* gBn = gB + k0 + 32;
            cp_async16(gAn + (size_t)c0r * K + c0k, &sAt[buf ^ 1][c0r * 32 + c0k]);
            cp_async16(gAn + (size_t)c1r * K + c1k, &sAt[buf ^ 1][c1r * 32 + c1k]);
            cp_async16(gBn + (size_t)c0r * K + c0k, &sBt[buf ^ 1][c0r * 32 + c0k]);
            cp_async16(gBn + (size_t)c1r * K + c1k, &sBt[buf ^ 1][c1r * 32 + c1k]);
        }

        // fragments from LDS
        Frag16 fa[2], fb[4];
        #pragma unroll
        for (int ti = 0; ti < 2; ++ti) {
            const __bf16* p = &sAt[buf][(wmBase + 16 * ti + lid) * 32];
            fa[ti].h[0] = *(const v8bf*)(p + kg * 8);
            fa[ti].h[1] = *(const v8bf*)(p + 16 + kg * 8);
        }
        #pragma unroll
        for (int tj = 0; tj < 4; ++tj) {
            const __bf16* p = &sBt[buf][(wnBase + 16 * tj + lid) * 32];
            fb[tj].h[0] = *(const v8bf*)(p + kg * 8);
            fb[tj].h[1] = *(const v8bf*)(p + 16 + kg * 8);
        }

        #pragma unroll
        for (int ti = 0; ti < 2; ++ti)
            #pragma unroll
            for (int tj = 0; tj < 4; ++tj)
                acc[ti][tj] = __builtin_amdgcn_wmma_f32_16x16x32_bf16(
                    false, fa[ti].v, false, fb[tj].v, (short)0, acc[ti][tj],
                    false, false);

        wait_async();
        __syncthreads();
        buf ^= 1;
    }

    // D 16x16 f32 layout: lanes 0-15 rows 0-7 (VGPR j -> row j), lanes 16-31 rows 8-15
    const int cn = lane & 15;
    const int ch = lane >> 4;
    #pragma unroll
    for (int ti = 0; ti < 2; ++ti) {
        #pragma unroll
        for (int tj = 0; tj < 4; ++tj) {
            #pragma unroll
            for (int j = 0; j < 8; ++j) {
                int row = m0 + wmBase + 16 * ti + ch * 8 + j;
                int col = n0 + wnBase + 16 * tj + cn;
                size_t idx = dOff + (size_t)row * N + col;
                float v = acc[ti][tj][j];
                if (MODE == 0) {
                    Df[idx] = v;
                } else if (MODE == 1) {
                    Db[idx] = (__bf16)v;
                } else if (MODE == 2) {
                    Df[idx] = v + bias[col] + resid[idx];
                } else if (MODE == 3) {
                    Db[idx] = (__bf16)gelu_f(v + bias[col]);
                } else {
                    Df[idx] = v + bias[col];
                }
            }
        }
    }
}

// ---------------------------------------------------------------------------
// Launcher
// ---------------------------------------------------------------------------
extern "C" void kernel_launch(void* const* d_in, const int* in_sizes, int n_in,
                              void* d_out, int out_size, void* d_ws, size_t ws_size,
                              hipStream_t stream) {
    const float* x       = (const float*)d_in[0];
    const float* ln1_g   = (const float*)d_in[1];
    const float* ln1_b   = (const float*)d_in[2];
    const float* w_attn  = (const float*)d_in[3];
    const float* b_attn  = (const float*)d_in[4];
    const float* w_proj  = (const float*)d_in[5];
    const float* b_proj  = (const float*)d_in[6];
    const float* ln2_g   = (const float*)d_in[7];
    const float* ln2_b   = (const float*)d_in[8];
    const float* w_fc    = (const float*)d_in[9];
    const float* b_fc    = (const float*)d_in[10];
    const float* w_mlp   = (const float*)d_in[11];
    const float* b_mlp   = (const float*)d_in[12];
    float* out = (float*)d_out;

    char* ws = (char*)d_ws;
    // region 0 (reused): qkv f32 (100.7MB) -> att f32 (64MB) + att bf16 -> m bf16 (67MB)
    const size_t OFF_R0    = 0;
    const size_t OFF_ATTBF = 67108864ull;                    // att_bf16 (33.5MB, inside region0)
    const size_t OFF_H1    = 100663296ull;                   // h1 bf16 (16MB) -> reused as y bf16
    const size_t OFF_Q     = OFF_H1 + 16777216ull;           // q bf16 -> reused as h2 bf16
    const size_t OFF_K     = OFF_Q + 16777216ull;            // k bf16
    const size_t OFF_VT    = OFF_K + 16777216ull;            // vT bf16
    const size_t OFF_X1    = OFF_VT + 16777216ull;           // x1 f32 (33.5MB)
    const size_t OFF_WATT  = OFF_X1 + 33554432ull;           // w_attn^T bf16 (6.3MB)
    const size_t OFF_WPRJ  = OFF_WATT + 6291456ull;          // w_proj^T bf16 (2.1MB)
    const size_t OFF_WFC   = OFF_WPRJ + 2097152ull;          // w_fc^T bf16 (8.4MB)
    const size_t OFF_W2    = OFF_WFC + 8388608ull;           // w_mlp_proj^T bf16 (8.4MB)

    float*  qkv_f  = (float*)(ws + OFF_R0);
    float*  att_f  = (float*)(ws + OFF_R0);
    __bf16* m_bf   = (__bf16*)(ws + OFF_R0);
    __bf16* att_bf = (__bf16*)(ws + OFF_ATTBF);
    __bf16* h1_bf  = (__bf16*)(ws + OFF_H1);
    __bf16* y_bf   = (__bf16*)(ws + OFF_H1);
    __bf16* q_bf   = (__bf16*)(ws + OFF_Q);
    __bf16* h2_bf  = (__bf16*)(ws + OFF_Q);
    __bf16* k_bf   = (__bf16*)(ws + OFF_K);
    __bf16* vT_bf  = (__bf16*)(ws + OFF_VT);
    float*  x1_f   = (float*)(ws + OFF_X1);
    __bf16* wattT  = (__bf16*)(ws + OFF_WATT);
    __bf16* wprjT  = (__bf16*)(ws + OFF_WPRJ);
    __bf16* wfcT   = (__bf16*)(ws + OFF_WFC);
    __bf16* w2T    = (__bf16*)(ws + OFF_W2);

    // --- weight convert + transpose ---
    wt_kernel<<<(CDIM * 3 * CDIM) / 256, 256, 0, stream>>>(w_attn, wattT, CDIM, 3 * CDIM);
    wt_kernel<<<(CDIM * CDIM) / 256, 256, 0, stream>>>(w_proj, wprjT, CDIM, CDIM);
    wt_kernel<<<(CDIM * HDIM) / 256, 256, 0, stream>>>(w_fc, wfcT, CDIM, HDIM);
    wt_kernel<<<(HDIM * CDIM) / 256, 256, 0, stream>>>(w_mlp, w2T, HDIM, CDIM);

    // --- LN1 ---
    ln_kernel<<<BT, 256, 0, stream>>>(x, ln1_g, ln1_b, h1_bf, CDIM);

    // --- qkv = h1 @ w_attn + b_attn : [8192, 3072] ---
    gemm_nt<4><<<dim3(3 * CDIM / 128, BT / 128, 1), 256, 0, stream>>>(
        h1_bf, wattT, qkv_f, nullptr, b_attn, nullptr,
        BT, 3 * CDIM, CDIM, 0, 0, 0);

    // --- split q (scaled), k, vT ---
    qkv_split_kernel<<<(BT * 3 * CDIM) / 256, 256, 0, stream>>>(qkv_f, q_bf, k_bf, vT_bf);

    // --- att = (q/sqrt(C)) @ k^T : batched [4][2048, 2048] ---
    gemm_nt<0><<<dim3(TDIM / 128, TDIM / 128, BDIM), 256, 0, stream>>>(
        q_bf, k_bf, att_f, nullptr, nullptr, nullptr,
        TDIM, TDIM, CDIM,
        (long long)TDIM * CDIM, (long long)TDIM * CDIM, (long long)TDIM * TDIM);

    // --- softmax rows -> bf16 ---
    softmax_kernel<<<BT, 256, 0, stream>>>(att_f, att_bf, TDIM);

    // --- y = att @ v : batched [4][2048, 1024] ---
    gemm_nt<1><<<dim3(CDIM / 128, TDIM / 128, BDIM), 256, 0, stream>>>(
        att_bf, vT_bf, nullptr, y_bf, nullptr, nullptr,
        TDIM, CDIM, TDIM,
        (long long)TDIM * TDIM, (long long)CDIM * TDIM, (long long)TDIM * CDIM);

    // --- x1 = x + y @ w_proj + b_proj ---
    gemm_nt<2><<<dim3(CDIM / 128, BT / 128, 1), 256, 0, stream>>>(
        y_bf, wprjT, x1_f, nullptr, b_proj, x,
        BT, CDIM, CDIM, 0, 0, 0);

    // --- LN2 ---
    ln_kernel<<<BT, 256, 0, stream>>>(x1_f, ln2_g, ln2_b, h2_bf, CDIM);

    // --- m = gelu(h2 @ w_fc + b_fc) : [8192, 4096] bf16 ---
    gemm_nt<3><<<dim3(HDIM / 128, BT / 128, 1), 256, 0, stream>>>(
        h2_bf, wfcT, nullptr, m_bf, b_fc, nullptr,
        BT, HDIM, CDIM, 0, 0, 0);

    // --- out = x1 + m @ w_mlp_proj + b_mlp_proj ---
    gemm_nt<2><<<dim3(CDIM / 128, BT / 128, 1), 256, 0, stream>>>(
        m_bf, w2T, out, nullptr, b_mlp, x1_f,
        BT, CDIM, HDIM, 0, 0, 0);
}
